// StagePolicyNetwork_12721693131094
// MI455X (gfx1250) — matrix operand order repaired and checked by reading, not compiled, not run on hardware
//
#include <hip/hip_runtime.h>
#include <hip/hip_bf16.h>

typedef __attribute__((ext_vector_type(16))) __bf16 v16bf;
typedef __attribute__((ext_vector_type(8)))  __bf16 v8bf;
typedef __attribute__((ext_vector_type(2)))  __bf16 v2bf;
typedef __attribute__((ext_vector_type(8)))  float  v8f;

#define NFEAT 5
#define DEMB  32
#define INPAD 128        // padded K: [0,32)=emb [32,64)=dag [64,96)=glob [96,101)=x [101,128)=0
#define WPB   8          // waves per block
#define TPB   (WPB * 32)
#define NEG_HUGE (-3.402823466e38f)

__device__ __forceinline__ void store_pk_bf16(__bf16* dst, float a, float b) {
    v2bf p; p[0] = (__bf16)a; p[1] = (__bf16)b;   // -> v_cvt_pk_bf16_f32 + ds_store_b32
    *(v2bf*)dst = p;
}

// ---------------------------------------------------------------------------
// Exclusive prefix sum of int64 counts -> int32 prefix (single block, 1024 thr)
// ---------------------------------------------------------------------------
__global__ __launch_bounds__(1024) void scan_kernel(
    const long long* __restrict__ cnt, int n, int* __restrict__ pfx)
{
    __shared__ int tmp[1024];
    __shared__ int carry;
    if (threadIdx.x == 0) carry = 0;
    __syncthreads();
    for (int base = 0; base < n; base += 1024) {
        int i = base + (int)threadIdx.x;
        int v = (i < n) ? (int)cnt[i] : 0;
        tmp[threadIdx.x] = v;
        __syncthreads();
        for (int off = 1; off < 1024; off <<= 1) {
            int t = (threadIdx.x >= (unsigned)off) ? tmp[threadIdx.x - off] : 0;
            __syncthreads();
            tmp[threadIdx.x] += t;
            __syncthreads();
        }
        if (i < n) pfx[i] = carry + tmp[threadIdx.x] - v;   // exclusive
        __syncthreads();
        if (threadIdx.x == 1023) carry += tmp[1023];
        __syncthreads();
    }
}

// ---------------------------------------------------------------------------
// Main fused gather + 4-layer MLP kernel. One wave per 16-node tile.
// Layers 1-3 use v_wmma_f32_16x16x32_bf16 (f32 accumulate).
// ---------------------------------------------------------------------------
__global__ __launch_bounds__(TPB) void mlp_score_kernel(
    const float* __restrict__ x,    const float* __restrict__ emb,
    const float* __restrict__ dagS, const float* __restrict__ globS,
    const int* __restrict__ dagPfx, const int* __restrict__ obsPfx,
    const unsigned char* __restrict__ mask,
    const float* __restrict__ W1, const float* __restrict__ b1,
    const float* __restrict__ W2, const float* __restrict__ b2,
    const float* __restrict__ W3, const float* __restrict__ b3,
    const float* __restrict__ W4, const float* __restrict__ b4,
    float* __restrict__ scores,
    int n, int nDags, int nObs, int numTiles)
{
    // Block-shared weights, N-major so B fragments are contiguous runs.
    // K is PERMUTED to match the staging layout: padded k -> original k:
    //   k < 96  : orig = k + 5   (emb 5..36 | dag 37..68 | glob 69..100)
    //   k < 101 : orig = k - 96  (x 0..4)
    //   else    : zero pad
    __shared__ __bf16 sW1[32 * 128];   // [n][k]
    __shared__ __bf16 sW2[16 * 32];    // [n][k]
    __shared__ __bf16 sW3[16 * 32];    // [n][k], padded (16x8 real)
    __shared__ float  sB1[32], sB2[16], sB3[8], sW4v[8], sB4[1];
    // Per-wave staging (separate arrays -> no aliasing hazards)
    __shared__ __bf16 stagBF[WPB][16 * INPAD];  // input tile, bf16
    __shared__ float  stagF [WPB][16 * 32];     // activation tile, f32
    __shared__ int    sIds  [WPB][32];          // dag ids [0..15], obs ids [16..31]

    const int tid = threadIdx.x;

    // ---- stage weights into LDS (transpose to N-major, permute+pad K) ----
    for (int t = tid; t < 32 * 128; t += TPB) {
        int nn = t >> 7, kk = t & 127;
        int ok = (kk < 96) ? (kk + 5) : (kk - 96);
        sW1[t] = (kk < 101) ? (__bf16)W1[ok * 32 + nn] : (__bf16)0.f;
    }
    for (int t = tid; t < 16 * 32; t += TPB) {
        int nn = t >> 5, kk = t & 31;
        sW2[t] = (__bf16)W2[kk * 16 + nn];
        sW3[t] = (kk < 16 && nn < 8) ? (__bf16)W3[kk * 8 + nn] : (__bf16)0.f;
    }
    if (tid < 32)      sB1[tid]      = b1[tid];
    else if (tid < 48) sB2[tid - 32] = b2[tid - 32];
    else if (tid < 56) sB3[tid - 48] = b3[tid - 48];
    else if (tid < 64) sW4v[tid - 56] = W4[tid - 56];
    else if (tid == 64) sB4[0] = b4[0];
    __syncthreads();

    const int lane = tid & 31;
    const int wv   = tid >> 5;
    const int m    = lane & 15;    // M-row (A) or N-col (B/C) index
    const int half = lane >> 4;
    __bf16* stag = stagBF[wv];
    float*  stf  = stagF[wv];
    int*    ids  = sIds[wv];

    // ---- zero the constant pad region (cols 101..127) ONCE per wave ----
    for (int t = lane; t < 16 * 27; t += 32) {
        int r = t / 27, c = 101 + t % 27;
        stag[(r << 7) + c] = (__bf16)0.f;
    }

    // ---- hoist B fragments (weights) into VGPRs, wave-invariant ----
    // B layout (32x16 bf16): lane%16 = N, element e -> K = 16*(lane/16)+e
    v16bf bw1[2][4];
#pragma unroll
    for (int t = 0; t < 2; ++t)
#pragma unroll
        for (int kb = 0; kb < 4; ++kb) {
            const __bf16* p = &sW1[((m + 16 * t) << 7) + 32 * kb + 16 * half];
            v8bf r0 = *(const v8bf*)p;
            v8bf r1 = *(const v8bf*)(p + 8);
            v16bf bb;
#pragma unroll
            for (int e = 0; e < 8; ++e) { bb[e] = r0[e]; bb[8 + e] = r1[e]; }
            bw1[t][kb] = bb;
        }
    v16bf bw2, bw3;
    {
        const __bf16* p = &sW2[(m << 5) + 16 * half];
        v8bf r0 = *(const v8bf*)p, r1 = *(const v8bf*)(p + 8);
#pragma unroll
        for (int e = 0; e < 8; ++e) { bw2[e] = r0[e]; bw2[8 + e] = r1[e]; }
        p = &sW3[(m << 5) + 16 * half];
        r0 = *(const v8bf*)p; r1 = *(const v8bf*)(p + 8);
#pragma unroll
        for (int e = 0; e < 8; ++e) { bw3[e] = r0[e]; bw3[8 + e] = r1[e]; }
    }

    const int gwave   = blockIdx.x * WPB + wv;
    const int gstride = gridDim.x * WPB;

    for (int tile = gwave; tile < numTiles; tile += gstride) {
        const int base = tile << 4;

        // ---- prefetch next tile's embedding rows (dominant HBM stream) ----
        {
            int nb = base + (gstride << 4);
            if (nb < n)
                __builtin_prefetch(&emb[(size_t)nb * DEMB + lane * 16], 0, 1);
        }

        // ---- ragged segment lookup: lanes 0-15 -> dag id, 16-31 -> obs id ----
        {
            const int* pfx = half ? obsPfx : dagPfx;
            const int  sz  = half ? nObs : nDags;
            int q = base + m; if (q >= n) q = n - 1;
            int lo = 0, hi = sz;
            while (hi - lo > 1) {
                int mid = (lo + hi) >> 1;
                bool le = pfx[mid] <= q;
                lo = le ? mid : lo;
                hi = le ? hi  : mid;
            }
            ids[lane] = lo;
        }

        // ---- branchless gather: 2 rows/pass, pair-packed bf16 stores ----
        // lane-half selects row parity, lane%16 selects column pair (cp, cp+1)
        const int cp = m << 1;
#pragma unroll
        for (int r2 = 0; r2 < 16; r2 += 2) {
            const int r = r2 + half;
            int node = base + r; if (node >= n) node = n - 1;
            const int dg = ids[r];
            const int ob = ids[16 + r];
            const float2 ev = *(const float2*)&emb[node * DEMB + cp];
            const float2 dv = *(const float2*)&dagS[dg * DEMB + cp];
            const float2 gv = *(const float2*)&globS[ob * DEMB + cp];
            __bf16* row = &stag[r << 7];
            store_pk_bf16(row + cp,      ev.x, ev.y);
            store_pk_bf16(row + 32 + cp, dv.x, dv.y);
            store_pk_bf16(row + 64 + cp, gv.x, gv.y);
        }
        // x features: 16 rows x 5 cols = 80 scalar elements
#pragma unroll
        for (int it = 0; it < 3; ++it) {
            int t = lane + 32 * it;
            if (t < 16 * NFEAT) {
                int r = t / NFEAT, f = t % NFEAT;
                int node = base + r; if (node >= n) node = n - 1;
                stag[(r << 7) + 96 + f] = (__bf16)x[node * NFEAT + f];
            }
        }

        // ---- layer 1: [16x128] @ [128x32] via 8 WMMAs ----
        // A layout (16x32 bf16): lane%16 = M, e -> K = 16*half+e
        v8f c0 = {}; v8f c1 = {};
#pragma unroll
        for (int kb = 0; kb < 4; ++kb) {
            const __bf16* p = &stag[(m << 7) + 32 * kb + 16 * half];
            v8bf r0 = *(const v8bf*)p;
            v8bf r1 = *(const v8bf*)(p + 8);
            v16bf a;
#pragma unroll
            for (int e = 0; e < 8; ++e) { a[e] = r0[e]; a[8 + e] = r1[e]; }
            c0 = __builtin_amdgcn_wmma_f32_16x16x32_bf16(false, a, false, bw1[0][kb],
                                                         (short)0, c0, false, false);
            c1 = __builtin_amdgcn_wmma_f32_16x16x32_bf16(false, a, false, bw1[1][kb],
                                                         (short)0, c1, false, false);
        }
        // C layout: lane%16 = N, reg r -> row M = r + 8*half
#pragma unroll
        for (int r = 0; r < 8; ++r) {
            int row = r + 8 * half;
            float v0 = c0[r] + sB1[m];      v0 = v0 > 0.f ? v0 : 0.f;
            float v1 = c1[r] + sB1[m + 16]; v1 = v1 > 0.f ? v1 : 0.f;
            stf[row * 32 + m]      = v0;
            stf[row * 32 + m + 16] = v1;
        }

        // ---- layer 2: [16x32] @ [32x16], one WMMA ----
        v16bf a2;
#pragma unroll
        for (int e = 0; e < 16; ++e) {
            int K = 16 * half + e;
            a2[e] = (__bf16)stf[(m << 5) + K];
        }
        v8f c2 = {};
        c2 = __builtin_amdgcn_wmma_f32_16x16x32_bf16(false, a2, false, bw2,
                                                     (short)0, c2, false, false);
#pragma unroll
        for (int r = 0; r < 8; ++r) {
            int row = r + 8 * half;
            float v = c2[r] + sB2[m]; v = v > 0.f ? v : 0.f;
            stf[row * 16 + m] = v;          // 16x16 tile (reuse stf)
        }

        // ---- layer 3: [16x16] @ [16x8] (K,N zero-padded), one WMMA ----
        v16bf a3;
#pragma unroll
        for (int e = 0; e < 16; ++e) {
            int K = 16 * half + e;
            a3[e] = (K < 16) ? (__bf16)stf[(m << 4) + K] : (__bf16)0.f;
        }
        v8f c3 = {};
        c3 = __builtin_amdgcn_wmma_f32_16x16x32_bf16(false, a3, false, bw3,
                                                     (short)0, c3, false, false);
        float bias3 = (m < 8) ? sB3[m] : 0.f;
#pragma unroll
        for (int r = 0; r < 8; ++r) {
            int row = r + 8 * half;
            float v = c3[r] + bias3; v = v > 0.f ? v : 0.f;
            stf[row * 16 + m] = (m < 8) ? v : 0.f;
        }

        // ---- layer 4 (8->1 dot) + mask + store, lanes 0..15 ----
        if (lane < 16) {
            int node = base + lane;
            if (node < n) {
                const float4 h0 = *(const float4*)&stf[lane * 16];
                const float4 h1 = *(const float4*)&stf[lane * 16 + 4];
                float s = sB4[0]
                        + h0.x * sW4v[0] + h0.y * sW4v[1]
                        + h0.z * sW4v[2] + h0.w * sW4v[3]
                        + h1.x * sW4v[4] + h1.y * sW4v[5]
                        + h1.z * sW4v[6] + h1.w * sW4v[7];
                scores[node] = mask[node] ? s : NEG_HUGE;
            }
        }
    }
}

// ---------------------------------------------------------------------------
// Deterministic softmax: block-tree max -> exp + partial sums -> scale
// ---------------------------------------------------------------------------
__global__ __launch_bounds__(256) void reduce_max_kernel(
    const float* __restrict__ s, int n, float* __restrict__ partial)
{
    __shared__ float red[256];
    float m = NEG_HUGE;
    for (int i = blockIdx.x * 256 + threadIdx.x; i < n; i += gridDim.x * 256)
        m = fmaxf(m, s[i]);
    red[threadIdx.x] = m; __syncthreads();
    for (int off = 128; off > 0; off >>= 1) {
        if (threadIdx.x < (unsigned)off)
            red[threadIdx.x] = fmaxf(red[threadIdx.x], red[threadIdx.x + off]);
        __syncthreads();
    }
    if (threadIdx.x == 0) partial[blockIdx.x] = red[0];
}

__global__ __launch_bounds__(256) void final_max_kernel(
    const float* __restrict__ partial, int np, float* __restrict__ outMax)
{
    __shared__ float red[256];
    float m = NEG_HUGE;
    for (int i = threadIdx.x; i < np; i += 256) m = fmaxf(m, partial[i]);
    red[threadIdx.x] = m; __syncthreads();
    for (int off = 128; off > 0; off >>= 1) {
        if (threadIdx.x < (unsigned)off)
            red[threadIdx.x] = fmaxf(red[threadIdx.x], red[threadIdx.x + off]);
        __syncthreads();
    }
    if (threadIdx.x == 0) *outMax = red[0];
}

__global__ __launch_bounds__(256) void exp_sum_kernel(
    const float* __restrict__ s, int n, const float* __restrict__ mx,
    float* __restrict__ outE, float* __restrict__ partial)
{
    __shared__ float red[256];
    float m = *mx;
    float acc = 0.f;
    for (int i = blockIdx.x * 256 + threadIdx.x; i < n; i += gridDim.x * 256) {
        float e = __expf(s[i] - m);
        outE[i] = e;
        acc += e;
    }
    red[threadIdx.x] = acc; __syncthreads();
    for (int off = 128; off > 0; off >>= 1) {
        if (threadIdx.x < (unsigned)off)
            red[threadIdx.x] += red[threadIdx.x + off];
        __syncthreads();
    }
    if (threadIdx.x == 0) partial[blockIdx.x] = red[0];
}

__global__ __launch_bounds__(256) void final_sum_kernel(
    const float* __restrict__ partial, int np, float* __restrict__ outInv)
{
    __shared__ float red[256];
    float acc = 0.f;
    for (int i = threadIdx.x; i < np; i += 256) acc += partial[i];
    red[threadIdx.x] = acc; __syncthreads();
    for (int off = 128; off > 0; off >>= 1) {
        if (threadIdx.x < (unsigned)off)
            red[threadIdx.x] += red[threadIdx.x + off];
        __syncthreads();
    }
    if (threadIdx.x == 0) *outInv = 1.f / red[0];
}

__global__ __launch_bounds__(256) void scale_kernel(
    float* __restrict__ out, int n, const float* __restrict__ inv)
{
    float s = *inv;
    for (int i = blockIdx.x * 256 + threadIdx.x; i < n; i += gridDim.x * 256)
        out[i] *= s;
}

// ---------------------------------------------------------------------------
static inline size_t align256(size_t v) { return (v + 255) & ~(size_t)255; }

extern "C" void kernel_launch(void* const* d_in, const int* in_sizes, int n_in,
                              void* d_out, int out_size, void* d_ws, size_t ws_size,
                              hipStream_t stream)
{
    const float*     x      = (const float*)d_in[0];
    const float*     emb    = (const float*)d_in[1];
    const float*     dagS   = (const float*)d_in[2];
    const float*     globS  = (const float*)d_in[3];
    const long long* cntDag = (const long long*)d_in[4];
    const long long* cntObs = (const long long*)d_in[5];
    const unsigned char* mask = (const unsigned char*)d_in[6];
    const float* W1 = (const float*)d_in[7];  const float* b1 = (const float*)d_in[8];
    const float* W2 = (const float*)d_in[9];  const float* b2 = (const float*)d_in[10];
    const float* W3 = (const float*)d_in[11]; const float* b3 = (const float*)d_in[12];
    const float* W4 = (const float*)d_in[13]; const float* b4 = (const float*)d_in[14];
    float* out = (float*)d_out;

    const int n     = in_sizes[6];   // stage_mask length == N_NODES
    const int nDags = in_sizes[4];
    const int nObs  = in_sizes[5];

    // workspace layout
    char* ws = (char*)d_ws;
    size_t o = 0;
    int*   dagPfx  = (int*)(ws + o); o += align256((size_t)nDags * 4);
    int*   obsPfx  = (int*)(ws + o); o += align256((size_t)nObs * 4);
    float* scores  = (float*)(ws + o); o += align256((size_t)n * 4);
    const int RBLK = 512;
    float* pMax    = (float*)(ws + o); o += align256((size_t)RBLK * 4);
    float* pSum    = (float*)(ws + o); o += align256((size_t)RBLK * 4);
    float* sMax    = (float*)(ws + o);
    float* sInv    = sMax + 1;

    // 1) ragged prefix sums
    scan_kernel<<<1, 1024, 0, stream>>>(cntDag, nDags, dagPfx);
    scan_kernel<<<1, 1024, 0, stream>>>(cntObs, nObs, obsPfx);

    // 2) fused gather + WMMA MLP -> masked scores
    const int numTiles = (n + 15) / 16;
    int gridMain = (numTiles + WPB - 1) / WPB;
    if (gridMain > 1280) gridMain = 1280;
    mlp_score_kernel<<<gridMain, TPB, 0, stream>>>(
        x, emb, dagS, globS, dagPfx, obsPfx, mask,
        W1, b1, W2, b2, W3, b3, W4, b4,
        scores, n, nDags, nObs, numTiles);

    // 3) softmax (deterministic tree reductions)
    reduce_max_kernel<<<RBLK, 256, 0, stream>>>(scores, n, pMax);
    final_max_kernel<<<1, 256, 0, stream>>>(pMax, RBLK, sMax);
    exp_sum_kernel<<<RBLK, 256, 0, stream>>>(scores, n, sMax, out, pSum);
    final_sum_kernel<<<1, 256, 0, stream>>>(pSum, RBLK, sInv);
    int gridScale = (n + 255) / 256;
    if (gridScale > 2048) gridScale = 2048;
    scale_kernel<<<gridScale, 256, 0, stream>>>(out, n, sInv);
}
